// SGMTSEncoder_25116968747560
// MI455X (gfx1250) — compile-verified, matrix operand order checked by reading, not run on hardware
//
#include <hip/hip_runtime.h>
#include <hip/hip_bf16.h>

#define DF     256
#define DSTATE 16
#define NP     196
#define NB     32
#define DPATCH 768
#define DHID   384
#define DLANG  896
#define ROWS   (NB * NP)   // 6272

typedef __attribute__((ext_vector_type(16))) __bf16 v16bf;
typedef __attribute__((ext_vector_type(8)))  float  v8f;

union __align__(32) FragBF { unsigned int u[8]; v16bf v; };
union __align__(32) FragF  { float f[8]; v8f v; };

__device__ __forceinline__ unsigned short f2bf(float f) {
  unsigned u = __float_as_uint(f);
  u += 0x7fffu + ((u >> 16) & 1u);          // round-to-nearest-even
  return (unsigned short)(u >> 16);
}
__device__ __forceinline__ float gelu_exact(float x) {
  return 0.5f * x * (1.f + erff(x * 0.70710678118654752440f));
}
__device__ __forceinline__ float softplus_f(float x) {
  return (x > 20.f) ? x : log1pf(expf(x));
}
__device__ __forceinline__ float waveRed(float v) {
#pragma unroll
  for (int o = 16; o > 0; o >>= 1) v += __shfl_down(v, o, 32);
  return v;
}
__device__ __forceinline__ float blockSum256(float v, float* red) {
  v = waveRed(v);
  const int wave = threadIdx.x >> 5, lane = threadIdx.x & 31;
  if (lane == 0) red[wave] = v;
  __syncthreads();
  float s = (threadIdx.x < 8) ? red[threadIdx.x] : 0.f;
  s += __shfl_down(s, 4, 32); s += __shfl_down(s, 2, 32); s += __shfl_down(s, 1, 32);
  if (threadIdx.x == 0) red[0] = s;
  __syncthreads();
  float out = red[0];
  __syncthreads();
  return out;
}

// ---------- weight prep ----------
__global__ __launch_bounds__(256) void k_transpose_bf16(const float* __restrict__ src,
    unsigned short* __restrict__ dst, int K, int N) {
  int idx = blockIdx.x * 256 + threadIdx.x;
  if (idx >= K * N) return;
  int n = idx / K, k = idx - n * K;
  dst[idx] = f2bf(src[(size_t)k * N + n]);   // dst[n][k] (N x K, row-major)
}
__global__ __launch_bounds__(256) void k_convert_bf16(const float* __restrict__ src,
    unsigned short* __restrict__ dst, int n) {
  int i = blockIdx.x * 256 + threadIdx.x;
  if (i < n) dst[i] = f2bf(src[i]);
}
__global__ void k_bcbias(const float* __restrict__ Bb, const float* __restrict__ Cb,
                         float* __restrict__ dst) {
  int i = threadIdx.x;
  if (i < 16) dst[i] = Bb[i];
  else if (i < 32) dst[i] = Cb[i - 16];
}

// ---------- patch extraction (fp32 -> bf16 rows of 768) ----------
__global__ __launch_bounds__(256) void k_patches(const float* __restrict__ img,
                                                 unsigned short* __restrict__ outp) {
  const int row = blockIdx.x, t = threadIdx.x;
  const int b = row / NP, u = row - b * NP;
  const int i = u / 14, j = u - i * 14;
  const int pi = t >> 4, pj = t & 15;
  const size_t ibase = ((size_t)b * 3) * 224 * 224 + (size_t)(i * 16 + pi) * 224 + (j * 16 + pj);
#pragma unroll
  for (int c = 0; c < 3; c++) {
    float v = img[ibase + (size_t)c * 224 * 224];
    outp[(size_t)row * DPATCH + c * 256 + t] = f2bf(v);
  }
}

// ---------- generic WMMA GEMM: C = A(bf16 MxK) * BT(bf16 NxK)^T + bias ----------
// block = 8 waves = (2 M-tiles) x (4 wave-cols of 32 cols); wave: 1x2 16x16 tiles
// EPI: 0 = +bias -> f32 ; 1 = gelu(+bias) -> bf16 ; 2 = softplus(+bias) -> f32
template <int EPI>
__global__ __launch_bounds__(256) void k_gemm(const unsigned short* __restrict__ A,
    const unsigned short* __restrict__ BT, const float* __restrict__ bias,
    float* __restrict__ outF, unsigned short* __restrict__ outH,
    int M, int N, int K) {
  const int wave = threadIdx.x >> 5, lane = threadIdx.x & 31;
  const int mtile = blockIdx.x * 2 + (wave >> 2);
  const int nbase = blockIdx.y * 128 + (wave & 3) * 32;
  if (mtile * 16 >= M || nbase >= N) return;
  const int r = lane & 15, hs = (lane >> 4) * 8;      // K half-select per 16-bit A/B layout
  const unsigned short* pa  = A  + (size_t)(mtile * 16 + r) * K + hs;
  const unsigned short* pb0 = BT + (size_t)(nbase + r) * K + hs;
  const unsigned short* pb1 = BT + (size_t)(nbase + 16 + r) * K + hs;
  FragF c0, c1;
#pragma unroll
  for (int i = 0; i < 8; i++) { c0.f[i] = 0.f; c1.f[i] = 0.f; }
  for (int k0 = 0; k0 < K; k0 += 32) {
    FragBF a, b0, b1;
    const uint4* qa = (const uint4*)(pa + k0);
    ((uint4*)a.u)[0] = qa[0];  ((uint4*)a.u)[1] = qa[2];      // K {hs..hs+7},{16+hs..}
    const uint4* q0 = (const uint4*)(pb0 + k0);
    ((uint4*)b0.u)[0] = q0[0]; ((uint4*)b0.u)[1] = q0[2];
    const uint4* q1 = (const uint4*)(pb1 + k0);
    ((uint4*)b1.u)[0] = q1[0]; ((uint4*)b1.u)[1] = q1[2];
    c0.v = __builtin_amdgcn_wmma_f32_16x16x32_bf16(false, a.v, false, b0.v,
                                                   (short)0, c0.v, false, false);
    c1.v = __builtin_amdgcn_wmma_f32_16x16x32_bf16(false, a.v, false, b1.v,
                                                   (short)0, c1.v, false, false);
  }
  const int mrow = mtile * 16 + (lane >> 4) * 8;   // C layout: VGPR i -> row mrow+i
  const int n0 = nbase + r, n1 = n0 + 16;
  const float bb0 = bias ? bias[n0] : 0.f;
  const float bb1 = bias ? bias[n1] : 0.f;
#pragma unroll
  for (int i = 0; i < 8; i++) {
    float v0 = c0.f[i] + bb0, v1 = c1.f[i] + bb1;
    size_t o0 = (size_t)(mrow + i) * N + n0, o1 = (size_t)(mrow + i) * N + n1;
    if (EPI == 0)      { outF[o0] = v0;                 outF[o1] = v1; }
    else if (EPI == 1) { outH[o0] = f2bf(gelu_exact(v0)); outH[o1] = f2bf(gelu_exact(v1)); }
    else               { outF[o0] = softplus_f(v0);     outF[o1] = softplus_f(v1); }
  }
}

// ---------- LayerNorm (in place) + row inv-norm for fn ----------
__global__ __launch_bounds__(256) void k_rowln_feats(float* __restrict__ f,
    const float* __restrict__ g, const float* __restrict__ b, float* __restrict__ invn) {
  __shared__ float red[8];
  const size_t row = blockIdx.x; const int t = threadIdx.x;
  float v = f[row * DF + t];
  float mean = blockSum256(v, red) * (1.f / DF);
  float d = v - mean;
  float var = blockSum256(d * d, red) * (1.f / DF);
  float y = d * rsqrtf(var + 1e-5f) * g[t] + b[t];
  f[row * DF + t] = y;
  float ss = blockSum256(y * y, red);
  if (t == 0) invn[row] = 1.f / fmaxf(sqrtf(ss), 1e-12f);
}

// ---------- gn = normalize(gate) ----------
__global__ __launch_bounds__(256) void k_rownorm(const float* __restrict__ src,
                                                 float* __restrict__ dst) {
  __shared__ float red[8];
  const size_t row = blockIdx.x; const int t = threadIdx.x;
  float v = src[row * DF + t];
  float ss = blockSum256(v * v, red);
  dst[row * DF + t] = v / fmaxf(sqrtf(ss), 1e-12f);
}

// ---------- sem + X = feats + sem * gprime (f32 + bf16) ----------
__global__ __launch_bounds__(256) void k_semX(const float* __restrict__ feats,
    const float* __restrict__ invn, const float* __restrict__ gn,
    const float* __restrict__ gprime, float* __restrict__ sem_o,
    float* __restrict__ X, unsigned short* __restrict__ Xbf) {
  __shared__ float red[8];
  const size_t row = blockIdx.x; const int b = (int)(row / NP); const int t = threadIdx.x;
  float fv = feats[row * DF + t];
  float dot = blockSum256(fv * gn[(size_t)b * DF + t], red);
  float sem = dot * invn[row];
  if (t == 0) sem_o[row] = sem;
  float x = fv + sem * gprime[(size_t)b * DF + t];
  X[row * DF + t] = x;
  Xbf[row * DF + t] = f2bf(x);
}

// ---------- per-batch MST (Kruskal, max) + BFS order ----------
__global__ __launch_bounds__(256) void k_tree(const float* __restrict__ feats,
    const float* __restrict__ invn, const float* __restrict__ sem_g,
    int* __restrict__ order_g, int* __restrict__ parent_g) {
  const int E = 364;                       // 14*13 horizontal + 13*14 vertical
  __shared__ float sem[NP];
  __shared__ float ew[512];
  __shared__ int   eid[512];
  __shared__ int   par[NP];
  __shared__ int   rnk[NP];
  __shared__ short adj[NP][4];
  __shared__ short deg[NP];
  __shared__ short q[NP];
  __shared__ unsigned char vis[NP];
  const int b = blockIdx.x, t = threadIdx.x;
  const int wave = t >> 5, lane = t & 31;
  for (int p = t; p < NP; p += 256) { sem[p] = sem_g[(size_t)b * NP + p]; vis[p] = 0; }
  for (int e = t; e < 512; e += 256) if (e >= E) { ew[e] = -3.0e38f; eid[e] = -1; }
  __syncthreads();
  for (int e = wave; e < E; e += 8) {
    int u, v;
    if (e < 182) { int i = e / 13, j = e - i * 13; u = i * 14 + j; v = u + 1; }
    else { int e2 = e - 182; int i = e2 / 14, j = e2 - i * 14; u = i * 14 + j; v = u + 14; }
    const float* fu = feats + ((size_t)b * NP + u) * DF;
    const float* fv = feats + ((size_t)b * NP + v) * DF;
    float s = 0.f;
    for (int k = lane; k < DF; k += 32) s += fu[k] * fv[k];
    s = waveRed(s);
    if (lane == 0) {
      ew[e]  = s * invn[(size_t)b * NP + u] * invn[(size_t)b * NP + v]
             + 0.5f * sem[u] * sem[v];
      eid[e] = (u << 16) | v;
    }
  }
  __syncthreads();
  // bitonic sort, descending by weight
  for (unsigned k = 2; k <= 512; k <<= 1)
    for (unsigned j = k >> 1; j > 0; j >>= 1) {
      for (unsigned i = t; i < 512; i += 256) {
        unsigned ixj = i ^ j;
        if (ixj > i) {
          bool desc = ((i & k) == 0);
          float a = ew[i], c = ew[ixj];
          bool sw = desc ? (a < c) : (a > c);
          if (sw) { ew[i] = c; ew[ixj] = a; int tm = eid[i]; eid[i] = eid[ixj]; eid[ixj] = tm; }
        }
      }
      __syncthreads();
    }
  if (t == 0) {
    for (int p = 0; p < NP; p++) { par[p] = p; rnk[p] = 0; deg[p] = 0; }
    int cnt = 0;
    for (int e = 0; e < E && cnt < NP - 1; e++) {
      int pk = eid[e]; if (pk < 0) continue;
      int u = pk >> 16, v = pk & 0xffff;
      int ru = u; while (par[ru] != ru) { par[ru] = par[par[ru]]; ru = par[ru]; }
      int rv = v; while (par[rv] != rv) { par[rv] = par[par[rv]]; rv = par[rv]; }
      if (ru != rv) {
        if (rnk[ru] < rnk[rv]) { int tm = ru; ru = rv; rv = tm; }
        par[rv] = ru;
        if (rnk[ru] == rnk[rv]) rnk[ru]++;
        adj[u][deg[u]++] = (short)v;
        adj[v][deg[v]++] = (short)u;
        cnt++;
      }
    }
    int root = 0; float best = sem[0];
    for (int p = 1; p < NP; p++) if (sem[p] > best) { best = sem[p]; root = p; }
    int head = 0, tail = 0;
    q[tail++] = (short)root; vis[root] = 1;
    parent_g[(size_t)b * NP + root] = root;
    int k = 0;
    while (head < tail) {
      int n = q[head++];
      order_g[(size_t)b * NP + k] = n; k++;
      for (int dg = 0; dg < deg[n]; dg++) {
        int nb = adj[n][dg];
        if (!vis[nb]) { vis[nb] = 1; parent_g[(size_t)b * NP + nb] = n; q[tail++] = (short)nb; }
      }
    }
  }
}

// ---------- serial tree scan: thread d owns feature dim d -> no barriers ----------
__global__ __launch_bounds__(256) void k_scan(const float* __restrict__ delta,
    const float* __restrict__ X, const float* __restrict__ BC,
    const float* __restrict__ A_log, const float* __restrict__ Dp,
    const int* __restrict__ order, const int* __restrict__ parent,
    float* __restrict__ h, unsigned short* __restrict__ ybf) {
  const int b = blockIdx.x, d = threadIdx.x;
  float A[DSTATE];
#pragma unroll
  for (int s = 0; s < DSTATE; s++) A[s] = -expf(A_log[d * DSTATE + s]);
  const float Dd = Dp[d];
  float* hb = h + (size_t)b * NP * (DF * DSTATE);
  const size_t base = (size_t)b * NP * DF;
  for (int step = 0; step < NP; step++) {
    const int node  = order[(size_t)b * NP + step];
    const int pnode = parent[(size_t)b * NP + node];
    const float dl = delta[base + (size_t)node * DF + d];
    const float x  = X[base + (size_t)node * DF + d];
    const float* bc = BC + ((size_t)b * NP + node) * 32;    // [0..15]=Bm [16..31]=Cm
    const float* hp = hb + (size_t)pnode * (DF * DSTATE) + d * DSTATE;
    float* hw = hb + (size_t)node * (DF * DSTATE) + d * DSTATE;
    const bool root = (pnode == node);
    float y = 0.f;
#pragma unroll
    for (int s = 0; s < DSTATE; s++) {
      float hpv = root ? 0.f : hp[s];       // same-thread RAW, kept in order by HW
      float v = expf(dl * A[s]) * hpv + dl * bc[s] * x;
      hw[s] = v;
      y += v * bc[16 + s];
    }
    y += Dd * x;
    ybf[base + (size_t)node * DF + d] = f2bf(y);
  }
}

// ---------- final LayerNorm -> d_out ----------
__global__ __launch_bounds__(256) void k_lnout(const float* __restrict__ z,
    const float* __restrict__ g, const float* __restrict__ b, float* __restrict__ outp) {
  __shared__ float red[8];
  const size_t row = blockIdx.x; const int t = threadIdx.x;
  float v = z[row * DF + t];
  float mean = blockSum256(v, red) * (1.f / DF);
  float d = v - mean;
  float var = blockSum256(d * d, red) * (1.f / DF);
  outp[row * DF + t] = d * rsqrtf(var + 1e-5f) * g[t] + b[t];
}

extern "C" void kernel_launch(void* const* d_in, const int* in_sizes, int n_in,
                              void* d_out, int out_size, void* d_ws, size_t ws_size,
                              hipStream_t stream) {
  const float* images  = (const float*)d_in[0];
  const float* g_task  = (const float*)d_in[1];
  const float* proj1_w = (const float*)d_in[2];
  const float* proj1_b = (const float*)d_in[3];
  const float* proj2_w = (const float*)d_in[4];
  const float* proj2_b = (const float*)d_in[5];
  const float* ln1_g   = (const float*)d_in[6];
  const float* ln1_b   = (const float*)d_in[7];
  const float* lgw     = (const float*)d_in[8];
  const float* wgp     = (const float*)d_in[9];
  const float* A_log   = (const float*)d_in[10];
  const float* Dp      = (const float*)d_in[11];
  const float* B_w     = (const float*)d_in[12];
  const float* B_b     = (const float*)d_in[13];
  const float* C_w     = (const float*)d_in[14];
  const float* C_b     = (const float*)d_in[15];
  const float* Wd_w    = (const float*)d_in[16];
  const float* Wd_b    = (const float*)d_in[17];
  const float* out_w   = (const float*)d_in[18];
  const float* out_b   = (const float*)d_in[19];
  const float* on_g    = (const float*)d_in[20];
  const float* on_b    = (const float*)d_in[21];
  (void)in_sizes; (void)n_in; (void)out_size; (void)ws_size;

  size_t off = 0;
  auto alloc = [&](size_t bytes) -> void* {
    off = (off + 255) & ~(size_t)255;
    void* p = (char*)d_ws + off;
    off += bytes;
    return p;
  };
  unsigned short* patches = (unsigned short*)alloc((size_t)ROWS * DPATCH * 2);
  unsigned short* w1t  = (unsigned short*)alloc((size_t)DHID * DPATCH * 2);
  unsigned short* hbuf = (unsigned short*)alloc((size_t)ROWS * DHID * 2);
  unsigned short* w2t  = (unsigned short*)alloc((size_t)DF * DHID * 2);
  float* fbuf = (float*)alloc((size_t)ROWS * DF * 4);
  float* invn = (float*)alloc((size_t)ROWS * 4);
  unsigned short* gbf  = (unsigned short*)alloc((size_t)NB * DLANG * 2);
  unsigned short* lgwT = (unsigned short*)alloc((size_t)DF * DLANG * 2);
  unsigned short* wgpT = (unsigned short*)alloc((size_t)DF * DLANG * 2);
  float* gate   = (float*)alloc((size_t)NB * DF * 4);
  float* gprime = (float*)alloc((size_t)NB * DF * 4);
  float* gn     = (float*)alloc((size_t)NB * DF * 4);
  float* semb   = (float*)alloc((size_t)ROWS * 4);
  float* Xf     = (float*)alloc((size_t)ROWS * DF * 4);
  unsigned short* Xbf = (unsigned short*)alloc((size_t)ROWS * DF * 2);
  unsigned short* wdT = (unsigned short*)alloc((size_t)DF * DF * 2);
  float* delta  = (float*)alloc((size_t)ROWS * DF * 4);
  unsigned short* bcT = (unsigned short*)alloc((size_t)32 * DF * 2);
  float* bcBias = (float*)alloc(32 * 4);
  float* BC     = (float*)alloc((size_t)ROWS * 32 * 4);
  int* order    = (int*)alloc((size_t)ROWS * 4);
  int* parentb  = (int*)alloc((size_t)ROWS * 4);
  float* hstate = (float*)alloc((size_t)ROWS * DF * DSTATE * 4);
  unsigned short* ybf  = (unsigned short*)alloc((size_t)ROWS * DF * 2);
  unsigned short* outT = (unsigned short*)alloc((size_t)DF * DF * 2);
  float* zbuf   = (float*)alloc((size_t)ROWS * DF * 4);

  auto tgrid = [](int n) { return dim3((unsigned)((n + 255) / 256)); };
  k_transpose_bf16<<<tgrid(DHID * DPATCH), 256, 0, stream>>>(proj1_w, w1t, DPATCH, DHID);
  k_transpose_bf16<<<tgrid(DF * DHID), 256, 0, stream>>>(proj2_w, w2t, DHID, DF);
  k_transpose_bf16<<<tgrid(DF * DLANG), 256, 0, stream>>>(lgw, lgwT, DLANG, DF);
  k_transpose_bf16<<<tgrid(DF * DLANG), 256, 0, stream>>>(wgp, wgpT, DLANG, DF);
  k_transpose_bf16<<<tgrid(DF * DF), 256, 0, stream>>>(Wd_w, wdT, DF, DF);
  k_transpose_bf16<<<tgrid(DF * DF), 256, 0, stream>>>(out_w, outT, DF, DF);
  k_transpose_bf16<<<tgrid(16 * DF), 256, 0, stream>>>(B_w, bcT, DF, 16);
  k_transpose_bf16<<<tgrid(16 * DF), 256, 0, stream>>>(C_w, bcT + 16 * DF, DF, 16);
  k_convert_bf16<<<tgrid(NB * DLANG), 256, 0, stream>>>(g_task, gbf, NB * DLANG);
  k_bcbias<<<1, 32, 0, stream>>>(B_b, C_b, bcBias);

  k_patches<<<ROWS, 256, 0, stream>>>(images, patches);

  auto ggrid = [](int M, int N) { return dim3((unsigned)((M + 31) / 32),
                                              (unsigned)((N + 127) / 128)); };
  // h = gelu(patches @ W1 + b1)  (bf16 out)
  k_gemm<1><<<ggrid(ROWS, DHID), 256, 0, stream>>>(patches, w1t, proj1_b,
                                                   nullptr, hbuf, ROWS, DHID, DPATCH);
  // f = h @ W2 + b2 ; LayerNorm in place + inv-norm
  k_gemm<0><<<ggrid(ROWS, DF), 256, 0, stream>>>(hbuf, w2t, proj2_b,
                                                 fbuf, nullptr, ROWS, DF, DHID);
  k_rowln_feats<<<ROWS, 256, 0, stream>>>(fbuf, ln1_g, ln1_b, invn);

  // gate / gprime / gn
  k_gemm<0><<<ggrid(NB, DF), 256, 0, stream>>>(gbf, lgwT, nullptr,
                                               gate, nullptr, NB, DF, DLANG);
  k_gemm<0><<<ggrid(NB, DF), 256, 0, stream>>>(gbf, wgpT, nullptr,
                                               gprime, nullptr, NB, DF, DLANG);
  k_rownorm<<<NB, 256, 0, stream>>>(gate, gn);

  // sem, X ; delta = softplus(X@Wd+b) ; [Bm|Cm] = X@[Bw|Cw]+b
  k_semX<<<ROWS, 256, 0, stream>>>(fbuf, invn, gn, gprime, semb, Xf, Xbf);
  k_gemm<2><<<ggrid(ROWS, DF), 256, 0, stream>>>(Xbf, wdT, Wd_b,
                                                 delta, nullptr, ROWS, DF, DF);
  k_gemm<0><<<ggrid(ROWS, 32), 256, 0, stream>>>(Xbf, bcT, bcBias,
                                                 BC, nullptr, ROWS, 32, DF);

  // tree build + serial scan
  k_tree<<<NB, 256, 0, stream>>>(fbuf, invn, semb, order, parentb);
  k_scan<<<NB, 256, 0, stream>>>(delta, Xf, BC, A_log, Dp, order, parentb, hstate, ybf);

  // z = y @ out_w + out_b ; final LayerNorm -> d_out
  k_gemm<0><<<ggrid(ROWS, DF), 256, 0, stream>>>(ybf, outT, out_b,
                                                 zbuf, nullptr, ROWS, DF, DF);
  k_lnout<<<ROWS, 256, 0, stream>>>(zbuf, on_g, on_b, (float*)d_out);
}
